// EdgeScoringNetwork_1795296330233
// MI455X (gfx1250) — matrix operand order repaired and checked by reading, not compiled, not run on hardware
//
#include <hip/hip_runtime.h>
#include <hip/hip_bf16.h>

#define FEATURE_DIM 128
#define EDGE_DIM 32
#define N_NODES 10000
#define N_EDGES 640000
#define POWER_ITERS 5
#define TEMPERATURE 0.66f
#define WAVES_PER_BLOCK 4
#define EDGES_PER_WAVE 16

typedef __attribute__((ext_vector_type(16))) _Float16 v16h;
typedef __attribute__((ext_vector_type(8)))  _Float16 hv8;
typedef __attribute__((ext_vector_type(4)))  _Float16 hv4;
typedef __attribute__((ext_vector_type(8)))  float    v8f;

// Pre-packed, spectral-normalized weights in WMMA B-fragment layout.
// B-fragment as packed here (KxN=32x16, contiguous k, same order as A): lanes 0-15 hold N=lane, K=0..15;
// lanes 16-31 hold N=lane-16, K=16..31; 16 halves per lane (8 VGPRs).
struct Params {
  _Float16 B1[8][2][32][16];   // layer1: [kchunk 0..7][ntile][lane][i] ; K = kc*32 + (lane/16)*16 + i
  _Float16 B2[2][32][16];      // layer2: K = (lane/16)*16 + i
  float w1_last[32];           // W1n[:,256] (distance column)
  float b1v[32];
  float b2v[32];
  float w3v[32];               // W3n
  float b3v;
};

__device__ __forceinline__ float block_reduce_sum(float x, float* red, int tid) {
  red[tid] = x;
  __syncthreads();
  for (int s = 128; s > 0; s >>= 1) {
    if (tid < s) red[tid] += red[tid + s];
    __syncthreads();
  }
  float r = red[0];
  __syncthreads();
  return r;
}

// Power iteration: returns sigma = ||W v_final|| (== u @ (W @ v) of the reference).
// u: R entries, v: C entries (shared), red: 256 entries (shared).
__device__ float spectral_sigma(const float* __restrict__ W, int R, int C,
                                float* u, float* v, float* red, int tid) {
  if (tid < R) u[tid] = 1.0f / sqrtf((float)R);
  __syncthreads();
  float sigma = 1.0f;
  for (int it = 0; it < POWER_ITERS; ++it) {
    // v = W^T u
    for (int j = tid; j < C; j += 256) {
      float s = 0.f;
      for (int i = 0; i < R; ++i) s += W[i * C + j] * u[i];
      v[j] = s;
    }
    __syncthreads();
    float part = 0.f;
    for (int j = tid; j < C; j += 256) part += v[j] * v[j];
    float nv = sqrtf(block_reduce_sum(part, red, tid)) + 1e-12f;
    for (int j = tid; j < C; j += 256) v[j] /= nv;
    __syncthreads();
    // u = W v
    float upart = 0.f;
    if (tid < R) {
      float s = 0.f;
      for (int j = 0; j < C; ++j) s += W[tid * C + j] * v[j];
      u[tid] = s;
      upart = s * s;
    }
    float nu = sqrtf(block_reduce_sum(upart, red, tid));
    sigma = nu;
    if (tid < R) u[tid] /= (nu + 1e-12f);
    __syncthreads();
  }
  return sigma;
}

__global__ __launch_bounds__(256) void prep_kernel(
    const float* __restrict__ W1, const float* __restrict__ pb1,
    const float* __restrict__ W2, const float* __restrict__ pb2,
    const float* __restrict__ W3, const float* __restrict__ pb3,
    Params* __restrict__ p) {
  __shared__ float sv[257];
  __shared__ float su[32];
  __shared__ float red[256];
  const int tid = threadIdx.x;

  const float sigma1 = spectral_sigma(W1, 32, 257, su, sv, red, tid);
  // pack B1 fragments (K = 0..255; distance column 256 handled in epilogue)
  for (int idx = tid; idx < 8 * 2 * 32 * 16; idx += 256) {
    const int i    = idx & 15;
    const int lane = (idx >> 4) & 31;
    const int nt   = (idx >> 9) & 1;
    const int kc   = idx >> 10;
    const int k = kc * 32 + (lane >> 4) * 16 + i;
    const int n = nt * 16 + (lane & 15);
    p->B1[kc][nt][lane][i] = (_Float16)(W1[n * 257 + k] / sigma1);
  }

  const float sigma2 = spectral_sigma(W2, 32, 32, su, sv, red, tid);
  for (int idx = tid; idx < 2 * 32 * 16; idx += 256) {
    const int i    = idx & 15;
    const int lane = (idx >> 4) & 31;
    const int nt   = idx >> 9;
    const int k = (lane >> 4) * 16 + i;
    const int n = nt * 16 + (lane & 15);
    p->B2[nt][lane][i] = (_Float16)(W2[n * 32 + k] / sigma2);
  }

  // W3 is 1x32: spectral norm == L2 norm of the row.
  float part = 0.f;
  for (int j = tid; j < 32; j += 256) part += W3[j] * W3[j];
  const float sigma3 = sqrtf(block_reduce_sum(part, red, tid));

  if (tid < 32) {
    p->w1_last[tid] = W1[tid * 257 + 256] / sigma1;
    p->b1v[tid] = pb1[tid];
    p->b2v[tid] = pb2[tid];
    p->w3v[tid] = W3[tid] / sigma3;
  }
  if (tid == 0) p->b3v = pb3[0];
}

__global__ __launch_bounds__(WAVES_PER_BLOCK * 32) void edge_kernel(
    const float* __restrict__ node_feat,
    const int* __restrict__ batch_ids,
    const int* __restrict__ src_nodes,
    const int* __restrict__ tgt_nodes,
    const Params* __restrict__ p,
    float* __restrict__ out) {
  // 264-half rows: 512B payload + 16B pad, keeps every 16B chunk aligned and skews banks.
  __shared__ alignas(16) _Float16 lds_feat[WAVES_PER_BLOCK][16][264];
  __shared__ alignas(16) _Float16 lds_h1[WAVES_PER_BLOCK][16][40];
  __shared__ float lds_dist[WAVES_PER_BLOCK][16];

  const int wave = threadIdx.x >> 5;
  const int lane = threadIdx.x & 31;
  const int edge_base = (blockIdx.x * WAVES_PER_BLOCK + wave) * EDGES_PER_WAVE;

  // ---- Gather: 2 lanes per edge, each covers 64 features of src and tgt ----
  {
    const int e_local = lane >> 1;
    const int half_id = lane & 1;
    const int e = edge_base + e_local;
    const int b = batch_ids[e];
    const int s = src_nodes[e];
    const int t = tgt_nodes[e];
    const float4* sp = (const float4*)(node_feat + ((size_t)b * N_NODES + (size_t)s) * FEATURE_DIM) + half_id * 16;
    const float4* tp = (const float4*)(node_feat + ((size_t)b * N_NODES + (size_t)t) * FEATURE_DIM) + half_id * 16;
    float d2 = 0.f;
#pragma unroll
    for (int c = 0; c < 16; ++c) {
      const float4 sv = sp[c];
      const float4 tv = tp[c];
      const float dx = sv.x - tv.x, dy = sv.y - tv.y, dz = sv.z - tv.z, dw = sv.w - tv.w;
      d2 += dx * dx + dy * dy + dz * dz + dw * dw;
      hv4 sh; sh[0] = (_Float16)sv.x; sh[1] = (_Float16)sv.y; sh[2] = (_Float16)sv.z; sh[3] = (_Float16)sv.w;
      hv4 th; th[0] = (_Float16)tv.x; th[1] = (_Float16)tv.y; th[2] = (_Float16)tv.z; th[3] = (_Float16)tv.w;
      const int col = half_id * 64 + c * 4;
      *(hv4*)&lds_feat[wave][e_local][col] = sh;           // src -> cols 0..127
      *(hv4*)&lds_feat[wave][e_local][128 + col] = th;     // tgt -> cols 128..255
    }
    d2 += __shfl_xor(d2, 1, 32);
    if (half_id == 0) lds_dist[wave][e_local] = sqrtf(d2);
  }
  __syncthreads();

  // ---- Layer 1: [16x256] @ [256x32] via 8 kchunks x 2 ntiles of 16x16x32 WMMA ----
  // A as loaded here (same k order as B1): lane holds row M=lane%16; halves 0..15 -> K = kc*32 + (lane/16)*16 + 0..15.
  // Two ds_load_b128 per fragment.
  const int m = lane & 15;
  const int koff = (lane >> 4) * 16;

  v8f acc0 = {};
  v8f acc1 = {};
#pragma unroll
  for (int kc = 0; kc < 8; ++kc) {
    union { v16h v; hv8 h[2]; } a;
    a.h[0] = *(const hv8*)&lds_feat[wave][m][kc * 32 + koff];
    a.h[1] = *(const hv8*)&lds_feat[wave][m][kc * 32 + koff + 8];
    const v16h b0 = *(const v16h*)&p->B1[kc][0][lane][0];
    const v16h b1f = *(const v16h*)&p->B1[kc][1][lane][0];
    acc0 = __builtin_amdgcn_wmma_f32_16x16x32_f16(false, a.v, false, b0, (short)0, acc0, false, false);
    acc1 = __builtin_amdgcn_wmma_f32_16x16x32_f16(false, a.v, false, b1f, (short)0, acc1, false, false);
  }

  // ---- Epilogue L1: + dist * w_last + b1, ReLU, restage as f16 A-input ----
  // C layout: VGPR r, lanes 0-15 -> M=r, lanes 16-31 -> M=8+r; N = lane%16 (+16 for acc1).
  {
    const int n0 = lane & 15;
    const float wl0 = p->w1_last[n0];
    const float wl1 = p->w1_last[n0 + 16];
    const float bb0 = p->b1v[n0];
    const float bb1 = p->b1v[n0 + 16];
#pragma unroll
    for (int r = 0; r < 8; ++r) {
      const int row = r + (lane >> 4) * 8;
      const float d = lds_dist[wave][row];
      const float h0 = fmaxf(acc0[r] + d * wl0 + bb0, 0.f);
      const float h1 = fmaxf(acc1[r] + d * wl1 + bb1, 0.f);
      lds_h1[wave][row][n0] = (_Float16)h0;
      lds_h1[wave][row][n0 + 16] = (_Float16)h1;
    }
  }
  __syncthreads();

  // ---- Layer 2: [16x32] @ [32x32] -> two WMMAs ----
  union { v16h v; hv8 h[2]; } a2;
  a2.h[0] = *(const hv8*)&lds_h1[wave][m][koff];
  a2.h[1] = *(const hv8*)&lds_h1[wave][m][koff + 8];
  const v16h b20 = *(const v16h*)&p->B2[0][lane][0];
  const v16h b21 = *(const v16h*)&p->B2[1][lane][0];
  v8f acc20 = {};
  v8f acc21 = {};
  acc20 = __builtin_amdgcn_wmma_f32_16x16x32_f16(false, a2.v, false, b20, (short)0, acc20, false, false);
  acc21 = __builtin_amdgcn_wmma_f32_16x16x32_f16(false, a2.v, false, b21, (short)0, acc21, false, false);

  // ---- Layer 3: +b2, ReLU, dot with w3 across N (butterfly over 16 lanes), output ----
  {
    const int n0 = lane & 15;
    const float w3a = p->w3v[n0];
    const float w3b = p->w3v[n0 + 16];
    const float bb20 = p->b2v[n0];
    const float bb21 = p->b2v[n0 + 16];
    float part[8];
#pragma unroll
    for (int r = 0; r < 8; ++r) {
      const float h0 = fmaxf(acc20[r] + bb20, 0.f);
      const float h1 = fmaxf(acc21[r] + bb21, 0.f);
      part[r] = h0 * w3a + h1 * w3b;
    }
#pragma unroll
    for (int mask = 1; mask < 16; mask <<= 1) {
#pragma unroll
      for (int r = 0; r < 8; ++r) part[r] += __shfl_xor(part[r], mask, 32);
    }
    const float b3 = p->b3v;
    if ((lane & 15) == 0) {
#pragma unroll
      for (int r = 0; r < 8; ++r) {
        const int row = r + (lane >> 4) * 8;
        float la = part[r] + b3;
        la = fminf(fmaxf(la, -5.f), 5.f);
        const float ew = 1.f / (1.f + __expf(-la * (1.0f / TEMPERATURE)));
        const int e = edge_base + row;
        out[e] = ew;                 // edge_weights
        out[N_EDGES + e] = la;       // logAlpha
      }
    }
  }
}

extern "C" void kernel_launch(void* const* d_in, const int* in_sizes, int n_in,
                              void* d_out, int out_size, void* d_ws, size_t ws_size,
                              hipStream_t stream) {
  const float* node_feat = (const float*)d_in[0];
  const float* W1 = (const float*)d_in[1];
  const float* b1 = (const float*)d_in[2];
  const float* W2 = (const float*)d_in[3];
  const float* b2 = (const float*)d_in[4];
  const float* W3 = (const float*)d_in[5];
  const float* b3 = (const float*)d_in[6];
  const int* batch_ids = (const int*)d_in[7];
  const int* src_nodes = (const int*)d_in[8];
  const int* tgt_nodes = (const int*)d_in[9];

  Params* p = (Params*)d_ws;

  prep_kernel<<<1, 256, 0, stream>>>(W1, b1, W2, b2, W3, b3, p);

  const int blocks = N_EDGES / (WAVES_PER_BLOCK * EDGES_PER_WAVE);  // 10000
  edge_kernel<<<blocks, WAVES_PER_BLOCK * 32, 0, stream>>>(
      node_feat, batch_ids, src_nodes, tgt_nodes, p, (float*)d_out);
}
